// SelfAttentionV3_81011673137528
// MI455X (gfx1250) — compile-verified, matrix operand order checked by reading, not run on hardware
//
#include <hip/hip_runtime.h>
#include <hip/hip_bf16.h>

// Problem constants (match reference)
#define TB  4
#define TT  2048
#define TD  1024
#define THD 16
#define TDK 64

// GEMM tiling: block 128x128, 8 waves (4 row x 2 col), wave tile 32x64.
#define BKT  32            // k-depth per LDS stage (1 WMMA k-step)
#define NKT  (TD / BKT)    // 32 stages
#define XSTR 48            // LDS row stride in f16 (96B: 16B-aligned)

typedef __attribute__((ext_vector_type(16))) _Float16 v16h;
typedef __attribute__((ext_vector_type(8)))  _Float16 v8h;
typedef __attribute__((ext_vector_type(8)))  float    v8f;

// ---------------------------------------------------------------------------
// CDNA5 async global->LDS copy (ASYNCcnt-tracked), with safe fallback
// ---------------------------------------------------------------------------
#if defined(__has_builtin)
#  if __has_builtin(__builtin_amdgcn_global_load_async_to_lds_b128)
#    define HAVE_ASYNC 1
#  endif
#endif
#ifndef HAVE_ASYNC
#  define HAVE_ASYNC 0
#endif

// Param type per hipcc diagnostic: vector_size(16) int, AS1 / AS3.
typedef int gv4i __attribute__((__vector_size__(4 * sizeof(int))));
typedef __attribute__((address_space(1))) gv4i* as1_v4i;
typedef __attribute__((address_space(3))) gv4i* as3_v4i;

__device__ __forceinline__ void cp_async16(_Float16* lds, const _Float16* g) {
#if HAVE_ASYNC
  // AS1 value = full VA; AS3 value = low 32 bits (flat->local is a trunc).
  __builtin_amdgcn_global_load_async_to_lds_b128(
      (as1_v4i)(unsigned long long)g,
      (as3_v4i)(unsigned int)(unsigned long long)lds, 0, 0);
#else
  (void)lds; (void)g;
#endif
}

__device__ __forceinline__ void wait_async0() {
#if HAVE_ASYNC
#  if __has_builtin(__builtin_amdgcn_s_wait_asynccnt)
  __builtin_amdgcn_s_wait_asynccnt(0);
#  else
  asm volatile("s_wait_asynccnt 0x0" ::: "memory");
#  endif
#endif
}

// D = A(16x32 f16) * B(32x16 f16) + C(16x16 f32)
__device__ __forceinline__ v8f wmma32(v16h a, v16h b, v8f c) {
  return __builtin_amdgcn_wmma_f32_16x16x32_f16(
      false, a, false, b, (short)0, c, false, false);
}

// 16x32 f16 A-fragment (or B-fragment of B^T) per ISA 7.12.2.
// lane L: row = tileRow + (L&15), h = L>>4. VGPR v K pair:
// v<4 -> K = 8h + 2v ; v>=4 -> K = 16 + 8h + 2(v-4).
// Per half-lane this is two contiguous 16B chunks -> 2x ds/global_load_b128.
__device__ __forceinline__ v16h frag_load(const _Float16* base, int ld, int row, int h) {
  v16h f;
  const _Float16* r = base + (long)row * ld;
#pragma unroll
  for (int v = 0; v < 8; ++v) {
    const int kb = (v < 4) ? (h * 8 + 2 * v) : (16 + h * 8 + 2 * (v - 4));
    f[2 * v]     = r[kb];
    f[2 * v + 1] = r[kb + 1];
  }
  return f;
}

// ---------------------------------------------------------------------------
// Double-buffered GEMM mainloop: block 128x128, wave tile 32x64.
// A panel gx [128 x TD], B^T panel gw [128 x TD], both ld = TD.
// rw0 = wave row base (0/32/64/96), cw0 = wave col base (0/64).
// acc[0..3] = rows rw0..rw0+15, acc[4..7] = rows rw0+16..rw0+31.
// Each B fragment feeds two WMMAs (1.5 ds_loads per WMMA).
// ---------------------------------------------------------------------------
__device__ __forceinline__ void gemm_mainloop(
    const _Float16* __restrict__ gx, const _Float16* __restrict__ gw,
    _Float16* sX /*2*128*XSTR*/, _Float16* sW /*2*128*XSTR*/,
    int tid, int rw0, int cw0, int ln, int h, v8f acc[8]) {
  const int xr0 = tid >> 2;          // 0..63
  const int xc0 = (tid & 3) * 8;     // 0,8,16,24 (16B chunks)

  auto stage_async = [&](int kt, int p) {
    const _Float16* gxk = gx + kt * BKT;
    const _Float16* gwk = gw + kt * BKT;
    _Float16* lx = sX + p * 128 * XSTR;
    _Float16* lw = sW + p * 128 * XSTR;
#pragma unroll
    for (int i = 0; i < 2; ++i) {
      cp_async16(lx + (xr0 + i * 64) * XSTR + xc0,
                 gxk + (long)(xr0 + i * 64) * TD + xc0);
      cp_async16(lw + (xr0 + i * 64) * XSTR + xc0,
                 gwk + (long)(xr0 + i * 64) * TD + xc0);
    }
  };

  auto compute = [&](int p) {
    const _Float16* bx = sX + p * 128 * XSTR;
    const _Float16* bw = sW + p * 128 * XSTR;
    v16h a0 = frag_load(bx, XSTR, rw0 + ln, h);
    v16h a1 = frag_load(bx, XSTR, rw0 + 16 + ln, h);
#pragma unroll
    for (int j = 0; j < 4; ++j) {
      v16h b = frag_load(bw, XSTR, cw0 + j * 16 + ln, h);
      acc[j]     = wmma32(a0, b, acc[j]);
      acc[4 + j] = wmma32(a1, b, acc[4 + j]);
    }
  };

#if HAVE_ASYNC
  stage_async(0, 0);
  wait_async0();
  __syncthreads();
  for (int kt = 0; kt < NKT; ++kt) {
    const int p = kt & 1;
    if (kt + 1 < NKT) stage_async(kt + 1, p ^ 1);  // overlaps with WMMAs below
    compute(p);
    wait_async0();
    __syncthreads();
  }
#else
  v8h rx[2], rw[2];
  auto load_regs = [&](int kt) {
    const _Float16* gxk = gx + kt * BKT;
    const _Float16* gwk = gw + kt * BKT;
#pragma unroll
    for (int i = 0; i < 2; ++i) {
      rx[i] = *(const v8h*)(gxk + (long)(xr0 + i * 64) * TD + xc0);
      rw[i] = *(const v8h*)(gwk + (long)(xr0 + i * 64) * TD + xc0);
    }
  };
  auto store_regs = [&](int p) {
    _Float16* lx = sX + p * 128 * XSTR;
    _Float16* lw = sW + p * 128 * XSTR;
#pragma unroll
    for (int i = 0; i < 2; ++i) {
      *(v8h*)(lx + (xr0 + i * 64) * XSTR + xc0) = rx[i];
      *(v8h*)(lw + (xr0 + i * 64) * XSTR + xc0) = rw[i];
    }
  };
  load_regs(0);
  store_regs(0);
  __syncthreads();
  for (int kt = 0; kt < NKT; ++kt) {
    const int p = kt & 1;
    if (kt + 1 < NKT) load_regs(kt + 1);   // global loads issued before compute
    compute(p);
    if (kt + 1 < NKT) store_regs(p ^ 1);   // loadcnt wait lands here, post-WMMA
    __syncthreads();
  }
#endif
}

// ---------------------------------------------------------------------------
// Stage 0: f32 -> f16 conversion
// ---------------------------------------------------------------------------
__global__ __launch_bounds__(256) void cvt_kernel(const float* __restrict__ s,
                                                  _Float16* __restrict__ d, int n) {
  int i = blockIdx.x * 256 + threadIdx.x;
  if (i < n) d[i] = (_Float16)s[i];
}

// ---------------------------------------------------------------------------
// Stage 1: fused QKV projection + bias + RoPE + layout scatter.
// Grid: x = 3*D/128 = 24, y = B*T/128 = 64. 256 threads (8 waves).
// Q,K -> [B][H][T][DK] f16; V -> [B][H][DK][T] f16 (transposed).
// ---------------------------------------------------------------------------
__global__ __launch_bounds__(256) void qkv_proj_kernel(
    const _Float16* __restrict__ xh, const _Float16* __restrict__ wqkv,
    const float* __restrict__ bq, const float* __restrict__ bk,
    const float* __restrict__ bv,
    _Float16* __restrict__ qo, _Float16* __restrict__ ko,
    _Float16* __restrict__ vT) {
  __shared__ _Float16 sX[2 * 128 * XSTR];
  __shared__ _Float16 sW[2 * 128 * XSTR];

  const int tid  = threadIdx.x;
  const int wave = tid >> 5, lane = tid & 31;
  const int h = lane >> 4, ln = lane & 15;
  const int rw0 = (wave >> 1) * 32;    // wave row base within block
  const int cw0 = (wave & 1) * 64;     // wave col base (one full head)
  const int mBase = blockIdx.y * 128;
  const int nBase = blockIdx.x * 128;  // [0, 3072), never straddles Q/K/V
  const int which = nBase >> 10;       // 0=Q 1=K 2=V
  const int eW    = (nBase & 1023) + cw0;  // head-aligned column base of wave
  const int head  = eW >> 6;
  const _Float16* W = wqkv + (long)which * TD * TD;

  v8f acc[8] = {};
  gemm_mainloop(xh + (long)mBase * TD, W + (long)(nBase & 1023) * TD,
                sX, sW, tid, rw0, cw0, ln, h, acc);

  const float* bias = (which == 0) ? bq : (which == 1) ? bk : bv;
  if (which < 2) {
    _Float16* out = (which == 0) ? qo : ko;
#pragma unroll
    for (int half = 0; half < 2; ++half) {
#pragma unroll
      for (int j = 0; j < 2; ++j) {
        const int i  = j * 16 + ln;                // rotation index; pair (i, i+32)
        const float b1 = bias[eW + i];
        const float b2 = bias[eW + i + 32];
        const float fr = expf(-(float)i * 0.28782313662425572f);  // 10000^(-i/32)
#pragma unroll
        for (int r = 0; r < 8; ++r) {
          const int m = mBase + rw0 + half * 16 + r + h * 8;
          const int bb = m >> 11;
          const int t  = m & (TT - 1);
          float sn, cs;
          sincosf((float)t * fr, &sn, &cs);
          const float x1 = acc[half * 4 + j][r] + b1;
          const float x2 = acc[half * 4 + j + 2][r] + b2;
          const long ob = (((long)bb * THD + head) * TT + t) * TDK;
          out[ob + i]      = (_Float16)(x1 * cs - x2 * sn);
          out[ob + i + 32] = (_Float16)(x1 * sn + x2 * cs);
        }
      }
    }
  } else {
#pragma unroll
    for (int half = 0; half < 2; ++half) {
#pragma unroll
      for (int j = 0; j < 4; ++j) {
        const int d  = j * 16 + ln;
        const float bbv = bias[eW + d];
#pragma unroll
        for (int r = 0; r < 8; ++r) {
          const int m = mBase + rw0 + half * 16 + r + h * 8;
          const int bb = m >> 11;
          const int t  = m & (TT - 1);
          vT[(((long)bb * THD + head) * TDK + d) * TT + t] =
              (_Float16)(acc[half * 4 + j][r] + bbv);
        }
      }
    }
  }
}

// ---------------------------------------------------------------------------
// Stage 2: wave-level flash attention (causal), one wave = 16 query rows.
// Grid: x = T/128 = 16, y = B*H = 64. No block barriers (per-wave LDS).
// ---------------------------------------------------------------------------
__global__ __launch_bounds__(256) void attn_kernel(
    const _Float16* __restrict__ q,   // [B][H][T][DK]
    const _Float16* __restrict__ k,   // [B][H][T][DK]
    const _Float16* __restrict__ vT,  // [B][H][DK][T]
    _Float16* __restrict__ o) {       // [B*T][D]
  __shared__ float    sS[8 * 16 * 33];
  __shared__ _Float16 sP[8 * 16 * 36];
  __shared__ float    sStat[8 * 16];

  const int tid  = threadIdx.x;
  const int wave = tid >> 5, lane = tid & 31;
  const int h = lane >> 4, ln = lane & 15;
  const int bh = blockIdx.y;
  const int bb = bh >> 4, hd = bh & 15;
  const int qBase = blockIdx.x * 128 + wave * 16;

  const _Float16* Q = q  + (long)bh * TT * TDK;
  const _Float16* K = k  + (long)bh * TT * TDK;
  const _Float16* V = vT + (long)bh * TDK * TT;
  float*    mySS   = &sS[wave * 16 * 33];
  _Float16* myP    = &sP[wave * 16 * 36];
  float*    myStat = &sStat[wave * 16];

  v16h aq0 = frag_load(Q, TDK, qBase + ln, h);
  v16h aq1 = frag_load(Q + 32, TDK, qBase + ln, h);

  v8f oacc[4] = {};
  float mrow = -1e30f, lrow = 0.f;
  const int kvEnd = qBase + 16;

  for (int kv0 = 0; kv0 < kvEnd; kv0 += 32) {
    v8f s[2] = {};
#pragma unroll
    for (int j = 0; j < 2; ++j) {
      int tk  = kv0 + j * 16 + ln;
      int tkc = tk < TT ? tk : (TT - 1);  // OOB cols are causally masked below
      v16h b0 = frag_load(K, TDK, tkc, h);
      s[j] = wmma32(aq0, b0, s[j]);
      v16h b1 = frag_load(K + 32, TDK, tkc, h);
      s[j] = wmma32(aq1, b1, s[j]);
    }
#pragma unroll
    for (int j = 0; j < 2; ++j)
#pragma unroll
      for (int r = 0; r < 8; ++r)
        mySS[(r + h * 8) * 33 + j * 16 + ln] = s[j][r];

    if (lane < 16) {
      const int tq = qBase + ln;
      float sv[32], tm = -1e30f;
#pragma unroll
      for (int n = 0; n < 32; ++n) {
        float val = mySS[ln * 33 + n] * 0.125f;  // 1/sqrt(64)
        if (kv0 + n > tq) val = -1e30f;
        sv[n] = val;
        tm = fmaxf(tm, val);
      }
      const float mnew = fmaxf(mrow, tm);
      const float corr = expf(mrow - mnew);
      float ssum = 0.f;
#pragma unroll
      for (int n = 0; n < 32; ++n) {
        float p = expf(sv[n] - mnew);
        ssum += p;
        myP[ln * 36 + n] = (_Float16)p;
      }
      lrow = lrow * corr + ssum;
      mrow = mnew;
      myStat[ln] = corr;
    }
#pragma unroll
    for (int j = 0; j < 4; ++j)
#pragma unroll
      for (int r = 0; r < 8; ++r) oacc[j][r] *= myStat[r + h * 8];

    v16h ap = frag_load(myP, 36, ln, h);
#pragma unroll
    for (int j = 0; j < 4; ++j) {
      v16h bv = frag_load(V + kv0, TT, j * 16 + ln, h);
      oacc[j] = wmma32(ap, bv, oacc[j]);
    }
  }

  if (lane < 16) myStat[ln] = 1.0f / lrow;
#pragma unroll
  for (int j = 0; j < 4; ++j) {
    const int d = j * 16 + ln;
#pragma unroll
    for (int r = 0; r < 8; ++r) {
      const int tq = qBase + r + h * 8;
      const float val = oacc[j][r] * myStat[r + h * 8];
      o[((long)bb * TT + tq) * TD + hd * TDK + d] = (_Float16)val;
    }
  }
}

// ---------------------------------------------------------------------------
// Stage 3: output projection  out = O @ Wo^T + bo  (f32 stores)
// Grid: x = D/128 = 8, y = B*T/128 = 64.
// ---------------------------------------------------------------------------
__global__ __launch_bounds__(256) void out_proj_kernel(
    const _Float16* __restrict__ oin, const _Float16* __restrict__ woh,
    const float* __restrict__ bo, float* __restrict__ out) {
  __shared__ _Float16 sX[2 * 128 * XSTR];
  __shared__ _Float16 sW[2 * 128 * XSTR];

  const int tid  = threadIdx.x;
  const int wave = tid >> 5, lane = tid & 31;
  const int h = lane >> 4, ln = lane & 15;
  const int rw0 = (wave >> 1) * 32;
  const int cw0 = (wave & 1) * 64;
  const int mBase = blockIdx.y * 128;
  const int nBase = blockIdx.x * 128;

  v8f acc[8] = {};
  gemm_mainloop(oin + (long)mBase * TD, woh + (long)nBase * TD,
                sX, sW, tid, rw0, cw0, ln, h, acc);

#pragma unroll
  for (int half = 0; half < 2; ++half) {
#pragma unroll
    for (int j = 0; j < 4; ++j) {
      const int n = nBase + cw0 + j * 16 + ln;
      const float bb = bo[n];
#pragma unroll
      for (int r = 0; r < 8; ++r) {
        const int m = mBase + rw0 + half * 16 + r + h * 8;
        out[(long)m * TD + n] = acc[half * 4 + j][r] + bb;
      }
    }
  }
}

// ---------------------------------------------------------------------------
extern "C" void kernel_launch(void* const* d_in, const int* in_sizes, int n_in,
                              void* d_out, int out_size, void* d_ws, size_t ws_size,
                              hipStream_t stream) {
  const float* x  = (const float*)d_in[0];
  const float* WQ = (const float*)d_in[1];
  const float* bq = (const float*)d_in[2];
  const float* WK = (const float*)d_in[3];
  const float* bk = (const float*)d_in[4];
  const float* WV = (const float*)d_in[5];
  const float* bv = (const float*)d_in[6];
  const float* WO = (const float*)d_in[7];
  const float* bo = (const float*)d_in[8];
  float* out = (float*)d_out;

  const long NX  = (long)TB * TT * TD;  // 8,388,608
  const long ND2 = (long)TD * TD;       // 1,048,576
  _Float16* ws   = (_Float16*)d_ws;
  _Float16* qb   = ws;                  // [B][H][T][DK]
  _Float16* kb   = qb + NX;             // [B][H][T][DK]
  _Float16* vTb  = kb + NX;             // [B][H][DK][T]
  _Float16* wqkv = vTb + NX;            // [3][E][D]
  _Float16* woh  = wqkv + 3 * ND2;      // [E][D]
  _Float16* xh   = woh + ND2;           // [B*T][D]; reused as attention output
  _Float16* ob   = xh;

  cvt_kernel<<<dim3((unsigned)((NX  + 255) / 256)), 256, 0, stream>>>(x,  xh,             (int)NX);
  cvt_kernel<<<dim3((unsigned)((ND2 + 255) / 256)), 256, 0, stream>>>(WQ, wqkv,           (int)ND2);
  cvt_kernel<<<dim3((unsigned)((ND2 + 255) / 256)), 256, 0, stream>>>(WK, wqkv + ND2,     (int)ND2);
  cvt_kernel<<<dim3((unsigned)((ND2 + 255) / 256)), 256, 0, stream>>>(WV, wqkv + 2 * ND2, (int)ND2);
  cvt_kernel<<<dim3((unsigned)((ND2 + 255) / 256)), 256, 0, stream>>>(WO, woh,            (int)ND2);

  qkv_proj_kernel<<<dim3(24, 64), 256, 0, stream>>>(xh, wqkv, bq, bk, bv, qb, kb, vTb);
  attn_kernel<<<dim3(16, 64), 256, 0, stream>>>(qb, kb, vTb, ob);
  out_proj_kernel<<<dim3(8, 64), 256, 0, stream>>>(ob, woh, bo, out);
}
